// _StreamGemmaAttention_19885698581155
// MI455X (gfx1250) — matrix operand order repaired and checked by reading, not compiled, not run on hardware
//
#include <hip/hip_runtime.h>
#include <cstdint>

// ---------------------------------------------------------------------------
// StreamGemmaAttention for MI455X (gfx1250, wave32)
//  - v_wmma_f32_16x16x32_f16 for all matrix math (f16 in, f32 accumulate)
//  - global_load_async_to_lds_b128 (ASYNCcnt) double-buffered GEMM staging
// ---------------------------------------------------------------------------

typedef __attribute__((ext_vector_type(16))) _Float16 v16h;
typedef __attribute__((ext_vector_type(8)))  float    v8f;

#define SEQ        4096
#define CHUNK      2048
#define START      2048
#define HIDDEN     2048
#define NHEADS     8
#define NKV        4
#define HDIM       256
#define QCOLS      (NHEADS * HDIM)   // 2048
#define KVCOLS     (NKV * HDIM)      // 1024
#define SOFTCAP    50.0f
#define SCALING    0.0625f           // 256^-0.5
#define NEG_BIG    (-1000000000.0f)

// ---------------- WMMA fragment loaders (ISA 7.12.2 layouts) ----------------
// A 16x32 f16: lane m=lane&15, khalf=lane>>4; VGPR v holds K = (v/4)*16 + khalf*8 + (v%4)*2 (+1)
__device__ __forceinline__ v16h frag_a(const _Float16* base, int stride,
                                       int mbase, int kbase, int lane) {
  union { v16h h; unsigned int u[8]; } f;
  const _Float16* p = base + (size_t)(mbase + (lane & 15)) * stride + kbase;
  const int khalf8 = ((lane >> 4) & 1) << 3;
#pragma unroll
  for (int v = 0; v < 8; ++v) {
    int k = ((v >> 2) << 4) + khalf8 + ((v & 3) << 1);
    f.u[v] = *(const unsigned int*)(p + k);
  }
  return f.h;
}

// B 32x16 f16: lane n=lane&15, khalf=lane>>4; VGPR v holds K = (v/4)*16 + khalf*8 + (v%4)*2 (+1)
__device__ __forceinline__ v16h frag_b(const _Float16* base, int stride,
                                       int nbase, int kbase, int lane) {
  union { v16h h; unsigned int u[8]; } f;
  const _Float16* p = base + (size_t)(nbase + (lane & 15)) * stride + kbase
                    + (((lane >> 4) & 1) << 3);
#pragma unroll
  for (int v = 0; v < 8; ++v) f.u[v] = *(const unsigned int*)(p + ((v >> 2) << 4) + ((v & 3) << 1));
  return f.h;
}

__device__ __forceinline__ v8f wmma_f16(v16h a, v16h b, v8f c) {
  return __builtin_amdgcn_wmma_f32_16x16x32_f16(false, a, false, b,
                                                (short)0, c, false, false);
}

// async global->LDS, 16B per lane, tracked by ASYNCcnt
__device__ __forceinline__ void async_b128(uint32_t lds_byte_off,
                                           const void* gaddr) {
  asm volatile("global_load_async_to_lds_b128 %0, %1, off"
               :: "v"(lds_byte_off), "v"(gaddr) : "memory");
}
__device__ __forceinline__ uint32_t lds_off(const void* p) {
  return (uint32_t)(uintptr_t)p;   // low 32 bits of generic = LDS offset
}

// ---------------------------- elementwise helpers ---------------------------
__global__ void cvt_f32_to_f16_kernel(const float* __restrict__ in,
                                      _Float16* __restrict__ out, int n) {
  int i = blockIdx.x * blockDim.x + threadIdx.x;
  if (i < n) out[i] = (_Float16)in[i];
}

// in: R x C f32 (row major)  ->  out: C x R f16 (row major)
__global__ __launch_bounds__(256)
void transpose_f16_kernel(const float* __restrict__ in,
                          _Float16* __restrict__ out, int R, int C) {
  __shared__ float tile[32][33];
  int bx = blockIdx.x * 32;
  int by = blockIdx.y * 32;
  int tx = threadIdx.x & 31;
  int ty = threadIdx.x >> 5;
#pragma unroll
  for (int i = 0; i < 32; i += 8)
    tile[ty + i][tx] = in[(size_t)(by + ty + i) * C + bx + tx];
  __syncthreads();
#pragma unroll
  for (int i = 0; i < 32; i += 8)
    out[(size_t)(bx + ty + i) * R + by + tx] = (_Float16)tile[tx][ty + i];
}

// ------------------------------ WMMA GEMM -----------------------------------
// C(MxN,f32) = A(MxK,f16 row major) * Bt(NxK,f16 row major == B col major)
// 128x128 block, K-step 32, 8 waves (4Mx2N), wave tile 32x64 = 2x4 WMMA tiles.
// Double-buffered LDS tiles filled by global_load_async_to_lds_b128:
//   tile = 128 rows x 32 f16 = 8 KB = 2 async b128 instructions per thread.
// Unpadded stride 32 keeps every 16B chunk 16B-aligned in LDS.
template <int N, int K>
__global__ __launch_bounds__(256)
void gemm_f16_kernel(const _Float16* __restrict__ A,
                     const _Float16* __restrict__ Bt,
                     float* __restrict__ C) {
  __shared__ __align__(16) _Float16 As[2][128 * 32];
  __shared__ __align__(16) _Float16 Bs[2][128 * 32];
  const int tid  = threadIdx.x;
  const int lane = tid & 31;
  const int wave = tid >> 5;
  const int wm = wave >> 1;               // 0..3 -> M offset wm*32
  const int wn = wave & 1;                // 0..1 -> N offset wn*64
  const int bm = blockIdx.y * 128;
  const int bn = blockIdx.x * 128;

  v8f acc[2][4];
#pragma unroll
  for (int i = 0; i < 2; ++i)
#pragma unroll
    for (int j = 0; j < 4; ++j) acc[i][j] = (v8f){};

  // each thread owns two 16B chunks per tile: chunk c -> row c/4, col (c%4)*8 f16
  const int cA0 = tid, cA1 = tid + 256;
  const int rA0 = cA0 >> 2, hA0 = (cA0 & 3) << 3;
  const int rA1 = cA1 >> 2, hA1 = (cA1 & 3) << 3;
  const _Float16* gA0 = A  + (size_t)(bm + rA0) * K + hA0;
  const _Float16* gA1 = A  + (size_t)(bm + rA1) * K + hA1;
  const _Float16* gB0 = Bt + (size_t)(bn + rA0) * K + hA0;
  const _Float16* gB1 = Bt + (size_t)(bn + rA1) * K + hA1;

  auto issue = [&](int k0, int buf) {
    async_b128(lds_off(&As[buf][rA0 * 32 + hA0]), gA0 + k0);
    async_b128(lds_off(&As[buf][rA1 * 32 + hA1]), gA1 + k0);
    async_b128(lds_off(&Bs[buf][rA0 * 32 + hA0]), gB0 + k0);
    async_b128(lds_off(&Bs[buf][rA1 * 32 + hA1]), gB1 + k0);
  };

  issue(0, 0);
  int buf = 0;
  for (int k0 = 0; k0 < K; k0 += 32) {
    if (k0 + 32 < K) {
      issue(k0 + 32, buf ^ 1);                       // prefetch next tile
      asm volatile("s_wait_asynccnt 4" ::: "memory"); // current tile landed
    } else {
      asm volatile("s_wait_asynccnt 0" ::: "memory");
    }
    __syncthreads();

    v16h bfr[4];
#pragma unroll
    for (int ns = 0; ns < 4; ++ns)
      bfr[ns] = frag_b(Bs[buf], 32, wn * 64 + ns * 16, 0, lane);
#pragma unroll
    for (int ms = 0; ms < 2; ++ms) {
      v16h a = frag_a(As[buf], 32, wm * 32 + ms * 16, 0, lane);
#pragma unroll
      for (int ns = 0; ns < 4; ++ns)
        acc[ms][ns] = wmma_f16(a, bfr[ns], acc[ms][ns]);
    }
    __syncthreads();
    buf ^= 1;
  }

  // C/D layout: lanes 0-15 M=r, lanes 16-31 M=r+8; N = lane&15
  const int hi8 = ((lane >> 4) & 1) << 3;
#pragma unroll
  for (int ms = 0; ms < 2; ++ms)
#pragma unroll
    for (int ns = 0; ns < 4; ++ns) {
      float* cp = C + (size_t)(bm + wm * 32 + ms * 16 + hi8) * N
                    + bn + wn * 64 + ns * 16 + (lane & 15);
#pragma unroll
      for (int r = 0; r < 8; ++r) cp[(size_t)r * N] = acc[ms][ns][r];
    }
}

// ------------------------ RMS-norm (+ optional RoPE) ------------------------
__global__ __launch_bounds__(256)
void norm_rope_kernel(const float* __restrict__ X, _Float16* __restrict__ Y,
                      const float* __restrict__ w,
                      const float* __restrict__ cosb,
                      const float* __restrict__ sinb,
                      int heads, int posOff, int doRope) {
  const int row  = blockIdx.x;
  const int head = blockIdx.y;
  const int t    = threadIdx.x;
  const int cols = heads * HDIM;
  const size_t base = (size_t)row * cols + head * HDIM;

  float v = X[base + t];
  __shared__ float red[256];
  __shared__ float xn[256];
  red[t] = v * v;
  __syncthreads();
#pragma unroll
  for (int s = 128; s > 0; s >>= 1) {
    if (t < s) red[t] += red[t + s];
    __syncthreads();
  }
  float inv = rsqrtf(red[0] * (1.0f / HDIM) + 1e-6f);
  float y = v * inv * (1.0f + w[t]);
  xn[t] = y;
  __syncthreads();

  float outv = y;
  if (doRope) {
    int pos = posOff + row;
    float c = cosb[(size_t)pos * HDIM + t];
    float s = sinb[(size_t)pos * HDIM + t];
    float rot = (t < 128) ? -xn[t + 128] : xn[t - 128];
    outv = y * c + rot * s;
  }
  Y[base + t] = (_Float16)outv;
}

// --------------------------- flash attention --------------------------------
#define QSTR 258            // 256 + 2 f16 pad
#define VSTR 66             // 64 + 2 f16 pad  (Vt[d][kp])
#define PSTR 66
#define SSTR 65             // f32

#define SM_QS   0
#define SM_KS   (64 * QSTR)
#define SM_VT   (SM_KS + 64 * QSTR)
#define SM_PS   (SM_VT + 256 * VSTR)
#define SM_H16  (SM_PS + 64 * PSTR)
#define ATTN_SMEM_BYTES (SM_H16 * 2 + (64 * SSTR + 3 * 64 + 512) * 4)

__global__ __launch_bounds__(256)
void attention_kernel(const _Float16* __restrict__ Q16,
                      const _Float16* __restrict__ K16,
                      const _Float16* __restrict__ V16,
                      float* __restrict__ AO) {
  extern __shared__ __align__(16) char smem[];
  _Float16* Qs = (_Float16*)smem + SM_QS;
  _Float16* Ks = (_Float16*)smem + SM_KS;
  _Float16* Vt = (_Float16*)smem + SM_VT;
  _Float16* Ps = (_Float16*)smem + SM_PS;
  float* Ss   = (float*)((_Float16*)smem + SM_H16);
  float* Mrow = Ss + 64 * SSTR;
  float* Lrow = Mrow + 64;
  float* Crow = Lrow + 64;
  float* PM   = Crow + 64;       // 256 partial maxima
  float* PSum = PM + 256;        // 256 partial sums

  const int qt = blockIdx.x;
  const int h  = blockIdx.y;
  const int hk = h >> 1;               // GQA: repeat 2
  const int tid  = threadIdx.x;
  const int lane = tid & 31;
  const int wave = tid >> 5;
  const int rg = wave >> 1;            // O row group 0..3
  const int ch = wave & 1;             // O col half
  const int hi8 = ((lane >> 4) & 1) << 3;
  const int qbase = qt * 64;

  for (int i = tid; i < 64 * 128; i += 256) {
    int r = i >> 7, dcol = (i & 127) << 1;
    *(uint32_t*)(Qs + r * QSTR + dcol) =
        *(const uint32_t*)(Q16 + (size_t)(qbase + r) * QCOLS + h * HDIM + dcol);
  }
  if (tid < 64) { Mrow[tid] = -1e30f; Lrow[tid] = 0.0f; }

  v8f accO[8];
#pragma unroll
  for (int t = 0; t < 8; ++t) accO[t] = (v8f){};

  const int ktiles = 33 + qt;          // causal: only tiles up to the diagonal
  for (int kt = 0; kt < ktiles; ++kt) {
    const int kbase = kt * 64;
    __syncthreads();

    for (int i = tid; i < 64 * 128; i += 256) {
      int r = i >> 7, dcol = (i & 127) << 1;
      *(uint32_t*)(Ks + r * QSTR + dcol) =
          *(const uint32_t*)(K16 + (size_t)(kbase + r) * KVCOLS + hk * HDIM + dcol);
    }
    for (int i = tid; i < 64 * 256; i += 256) {
      int d = i & 255, kp = i >> 8;
      Vt[d * VSTR + kp] = V16[(size_t)(kbase + kp) * KVCOLS + hk * HDIM + d];
    }
    __syncthreads();

    // ---- S = Q K^T (64x64): 16 tiles, 2 per wave, 8 K-steps over d
    for (int i = 0; i < 2; ++i) {
      int t = wave * 2 + i;
      int mt = t >> 2, nt = t & 3;
      v8f s = (v8f){};
#pragma unroll
      for (int ks = 0; ks < 8; ++ks) {
        v16h a = frag_a(Qs, QSTR, mt * 16, ks * 32, lane);
        v16h b = frag_b(Ks, QSTR, nt * 16, ks * 32, lane);
        s = wmma_f16(a, b, s);
      }
#pragma unroll
      for (int r = 0; r < 8; ++r) {
        int m = mt * 16 + r + hi8;
        int n = nt * 16 + (lane & 15);
        int qa = START + qbase + m;
        int ka = kbase + n;
        float sv = s[r] * SCALING;
        sv = tanhf(sv * (1.0f / SOFTCAP)) * SOFTCAP;
        sv += (ka > qa) ? NEG_BIG : 0.0f;
        Ss[m * SSTR + n] = sv;
      }
    }
    __syncthreads();

    // ---- online softmax: 4 threads per row, 16 elements each
    {
      const int r  = tid >> 2;
      const int qv = tid & 3;
      const float* srow = Ss + r * SSTR + qv * 16;
      float mx = -1e30f;
#pragma unroll
      for (int j = 0; j < 16; ++j) mx = fmaxf(mx, srow[j]);
      PM[tid] = mx;
      __syncthreads();
      float mo = Mrow[r];
      float mnew = fmaxf(mo, fmaxf(fmaxf(PM[r * 4 + 0], PM[r * 4 + 1]),
                                   fmaxf(PM[r * 4 + 2], PM[r * 4 + 3])));
      float sum = 0.0f;
      _Float16* prow = Ps + r * PSTR + qv * 16;
#pragma unroll
      for (int j = 0; j < 16; ++j) {
        float p = __expf(srow[j] - mnew);
        sum += p;
        prow[j] = (_Float16)p;
      }
      PSum[tid] = sum;
      __syncthreads();
      if (qv == 0) {
        float s4 = PSum[tid] + PSum[tid + 1] + PSum[tid + 2] + PSum[tid + 3];
        float corr = __expf(mo - mnew);
        Mrow[r] = mnew;
        Lrow[r] = Lrow[r] * corr + s4;
        Crow[r] = corr;
      }
    }
    __syncthreads();

    // ---- rescale O, then O += P * V (each wave: 16x128 slab)
#pragma unroll
    for (int t = 0; t < 8; ++t)
#pragma unroll
      for (int r = 0; r < 8; ++r)
        accO[t][r] *= Crow[rg * 16 + r + hi8];
#pragma unroll
    for (int ks = 0; ks < 2; ++ks) {
      v16h a = frag_a(Ps, PSTR, rg * 16, ks * 32, lane);
#pragma unroll
      for (int t = 0; t < 8; ++t) {
        v16h b = frag_b(Vt, VSTR, ch * 128 + t * 16, ks * 32, lane);
        accO[t] = wmma_f16(a, b, accO[t]);
      }
    }
  }
  __syncthreads();

#pragma unroll
  for (int t = 0; t < 8; ++t)
#pragma unroll
    for (int r = 0; r < 8; ++r) {
      int m = rg * 16 + r + hi8;
      int col = ch * 128 + t * 16 + (lane & 15);
      AO[(size_t)(qbase + m) * QCOLS + h * HDIM + col] = accO[t][r] / Lrow[m];
    }
}

// ------------------------------- launcher -----------------------------------
extern "C" void kernel_launch(void* const* d_in, const int* in_sizes, int n_in,
                              void* d_out, int out_size, void* d_ws, size_t ws_size,
                              hipStream_t stream) {
  const float* hidden = (const float*)d_in[0];
  const float* cosb   = (const float*)d_in[1];
  const float* sinb   = (const float*)d_in[2];
  const float* wq = (const float*)d_in[4];
  const float* wk = (const float*)d_in[5];
  const float* wv = (const float*)d_in[6];
  const float* wo = (const float*)d_in[7];
  const float* qw = (const float*)d_in[8];
  const float* kw = (const float*)d_in[9];
  const float* vw = (const float*)d_in[10];
  float* out = (float*)d_out;

  char* ws = (char*)d_ws;
  size_t off = 0;
  auto carve = [&](size_t bytes) -> char* {
    char* p = ws + off;
    off += (bytes + 255) & ~(size_t)255;
    return p;
  };
  _Float16* H16  = (_Float16*)carve((size_t)SEQ * HIDDEN * 2);
  _Float16* WQt  = (_Float16*)carve((size_t)QCOLS * HIDDEN * 2);
  _Float16* WKt  = (_Float16*)carve((size_t)KVCOLS * HIDDEN * 2);
  _Float16* WVt  = (_Float16*)carve((size_t)KVCOLS * HIDDEN * 2);
  _Float16* WOt  = (_Float16*)carve((size_t)HIDDEN * QCOLS * 2);
  float*    Qf   = (float*)carve((size_t)CHUNK * QCOLS * 4);
  float*    Kf   = (float*)carve((size_t)SEQ * KVCOLS * 4);
  float*    Vf   = (float*)carve((size_t)SEQ * KVCOLS * 4);
  _Float16* Q16  = (_Float16*)carve((size_t)CHUNK * QCOLS * 2);
  _Float16* K16  = (_Float16*)carve((size_t)SEQ * KVCOLS * 2);
  _Float16* V16  = (_Float16*)carve((size_t)SEQ * KVCOLS * 2);
  float*    AO   = (float*)carve((size_t)CHUNK * QCOLS * 4);
  _Float16* AO16 = (_Float16*)carve((size_t)CHUNK * QCOLS * 2);

  cvt_f32_to_f16_kernel<<<(SEQ * HIDDEN) / 256, 256, 0, stream>>>(hidden, H16, SEQ * HIDDEN);
  transpose_f16_kernel<<<dim3(QCOLS / 32, HIDDEN / 32), 256, 0, stream>>>(wq, WQt, HIDDEN, QCOLS);
  transpose_f16_kernel<<<dim3(KVCOLS / 32, HIDDEN / 32), 256, 0, stream>>>(wk, WKt, HIDDEN, KVCOLS);
  transpose_f16_kernel<<<dim3(KVCOLS / 32, HIDDEN / 32), 256, 0, stream>>>(wv, WVt, HIDDEN, KVCOLS);
  transpose_f16_kernel<<<dim3(HIDDEN / 32, QCOLS / 32), 256, 0, stream>>>(wo, WOt, QCOLS, HIDDEN);

  gemm_f16_kernel<QCOLS, HIDDEN><<<dim3(QCOLS / 128, CHUNK / 128), 256, 0, stream>>>(
      H16 + (size_t)START * HIDDEN, WQt, Qf);
  gemm_f16_kernel<KVCOLS, HIDDEN><<<dim3(KVCOLS / 128, SEQ / 128), 256, 0, stream>>>(
      H16, WKt, Kf);
  gemm_f16_kernel<KVCOLS, HIDDEN><<<dim3(KVCOLS / 128, SEQ / 128), 256, 0, stream>>>(
      H16, WVt, Vf);

  norm_rope_kernel<<<dim3(CHUNK, NHEADS), 256, 0, stream>>>(Qf, Q16, qw, cosb, sinb, NHEADS, START, 1);
  norm_rope_kernel<<<dim3(SEQ, NKV), 256, 0, stream>>>(Kf, K16, kw, cosb, sinb, NKV, 0, 1);
  norm_rope_kernel<<<dim3(SEQ, NKV), 256, 0, stream>>>(Vf, V16, vw, cosb, sinb, NKV, 0, 0);

  (void)hipFuncSetAttribute((const void*)attention_kernel,
                            hipFuncAttributeMaxDynamicSharedMemorySize,
                            ATTN_SMEM_BYTES);
  attention_kernel<<<dim3(CHUNK / 64, NHEADS), 256, ATTN_SMEM_BYTES, stream>>>(Q16, K16, V16, AO);

  cvt_f32_to_f16_kernel<<<(CHUNK * QCOLS) / 256, 256, 0, stream>>>(AO, AO16, CHUNK * QCOLS);
  gemm_f16_kernel<QCOLS, QCOLS><<<dim3(HIDDEN / 128, CHUNK / 128), 256, 0, stream>>>(
      AO16, WOt, out);

  (void)in_sizes; (void)n_in; (void)out_size; (void)ws_size;
}
